// Attention_6236292514476
// MI455X (gfx1250) — matrix operand order repaired and checked by reading, not compiled, not run on hardware
//
#include <hip/hip_runtime.h>
#include <hip/hip_bf16.h>

// ---------------------------------------------------------------------------
// Causal multi-head attention forward for MI455X (gfx1250, wave32, WMMA).
// B=2, S=2048, D_MODEL=1024, H=16, D_HEAD=64.  f16 operands, f32 accumulate.
// Uses v_wmma_f32_16x16x32_f16 + GLOBAL_LOAD_ASYNC_TO_LDS_B128 staging.
// ---------------------------------------------------------------------------

#define SEQ    2048
#define DM     1024
#define HEADS  16
#define DH     64
#define IGN    (-1e30f)

typedef __attribute__((ext_vector_type(16))) _Float16 v16h;
typedef __attribute__((ext_vector_type(8)))  float    v8f;

union FragU { unsigned u[8]; v16h h; };          // one 16x32 f16 WMMA operand
union HU   { _Float16 h; unsigned short s; };

static __device__ inline unsigned short f2h(float x) {
    HU t; t.h = (_Float16)x; return t.s;
}

static __device__ inline v8f wmma16(const FragU& a, const FragU& b, v8f c) {
    // v_wmma_f32_16x16x32_f16  D = A(16x32) * B(32x16) + C
    return __builtin_amdgcn_wmma_f32_16x16x32_f16(
        /*neg_a=*/false, a.h, /*neg_b=*/false, b.h,
        /*c_mod=*/(short)0, c, /*reuse_a=*/false, /*reuse_b=*/false);
}

// Async DMA: 16 bytes global -> LDS per lane (GVS addressing, ASYNCcnt).
// lds  = LDS byte address (low 32 bits of generic shared pointer)
// base = uniform 64-bit global base (SGPR pair), voff = per-lane byte offset
static __device__ inline void async_ld_b128(unsigned lds, const void* base,
                                            unsigned voff) {
    asm volatile("global_load_async_to_lds_b128 %0, %1, %2"
                 :: "v"(lds), "v"(voff), "s"(base) : "memory");
}
static __device__ inline void wait_async0() {
    asm volatile("s_wait_asynccnt 0x0" ::: "memory");
}
static __device__ inline unsigned lds_addr(const void* p) {
    return (unsigned)(unsigned long long)p;       // LDS address is the pointer's low 32 bits
}

// row reductions across the 16-lane half-wave holding one C-matrix row
static __device__ inline float rowmax16(float v) {
    v = fmaxf(v, __shfl_xor(v, 1, 32));
    v = fmaxf(v, __shfl_xor(v, 2, 32));
    v = fmaxf(v, __shfl_xor(v, 4, 32));
    v = fmaxf(v, __shfl_xor(v, 8, 32));
    return v;
}
static __device__ inline float rowsum16(float v) {
    v += __shfl_xor(v, 1, 32);
    v += __shfl_xor(v, 2, 32);
    v += __shfl_xor(v, 4, 32);
    v += __shfl_xor(v, 8, 32);
    return v;
}

// ---------------------------------------------------------------------------
// Kernel 1: fused QKV projection.
// C[4096, 3072] = X[4096,1024] * [Wq|Wk|Wv]  (+bias), emitted as f16.
// q: head-major [b][h][s][dh], pre-scaled by 1/sqrt(64)
// k: head-major [b][h][s][dh]
// v: head-TRANSPOSED [b][h][dh][s]  (so kernel 2 can async-DMA it straight in)
// Block: 256 thr (8 waves, 4x2 tiles), block tile 64(M) x 32(N), K-step 32.
// ---------------------------------------------------------------------------
__global__ __launch_bounds__(256)
void qkv_kernel(const float* __restrict__ x,
                const float* __restrict__ Wq, const float* __restrict__ Wk,
                const float* __restrict__ Wv,
                const float* __restrict__ bq, const float* __restrict__ bk,
                const float* __restrict__ bv,
                unsigned short* __restrict__ qh,
                unsigned short* __restrict__ kh,
                unsigned short* __restrict__ vhT)
{
    __shared__ __align__(16) unsigned short sA[64 * 32];   // [row][k]
    __shared__ __align__(16) unsigned short sB[32 * 32];   // [n][k] transposed

    const int t    = threadIdx.x;
    const int lane = t & 31;
    const int w    = t >> 5;
    const int wm   = w >> 1;        // 0..3
    const int wn   = w & 1;         // 0..1
    const int hi   = lane >> 4;     // 0/1
    const int ln   = lane & 15;

    const int R0 = blockIdx.y * 64;        // row tile (b*2048+s)
    const int C0 = blockIdx.x * 32;        // col in [0,3072)
    const int region = C0 >> 10;           // 0=Q 1=K 2=V
    const int local  = C0 & 1023;
    const int hd  = local >> 6;            // head (constant across tile)
    const int kk0 = local & 63;            // 0 or 32

    const float* W    = (region == 0) ? Wq : (region == 1 ? Wk : Wv);
    const float* bias = (region == 0) ? bq : (region == 1 ? bk : bv);
    const float* Wbase = W + (size_t)hd * (DM * DH) + kk0;

    v8f acc = {};

    for (int kd0 = 0; kd0 < DM; kd0 += 32) {
        // stage A: 64x32 f32 -> f16, row-major
        #pragma unroll
        for (int i = 0; i < 8; ++i) {
            int idx = t + 256 * i;
            int r = idx >> 5, c = idx & 31;
            sA[r * 32 + c] = f2h(x[(size_t)(R0 + r) * DM + kd0 + c]);
        }
        // stage B: 32(K) x 32(N) f32 -> f16, transposed [n][k]
        #pragma unroll
        for (int i = 0; i < 8; ++i) {
            int idx = t + 256 * i;
            int dk = idx >> 5, n = idx & 31;
            sB[n * 32 + dk] = f2h(Wbase[(size_t)(kd0 + dk) * DH + n]);
        }
        __syncthreads();

        FragU a, bf;
        const unsigned* pA = (const unsigned*)sA;
        const unsigned* pB = (const unsigned*)sB;
        #pragma unroll
        for (int v = 0; v < 8; ++v) {
            int ka = hi * 8 + ((v < 4) ? 2 * v : 2 * v + 8);   // A layout
            a.u[v]  = pA[(wm * 16 + ln) * 16 + (ka >> 1)];
            int kb = hi * 8 + ((v < 4) ? 2 * v : 2 * v + 8);   // B layout
            bf.u[v] = pB[(wn * 16 + ln) * 16 + (kb >> 1)];
        }
        acc = wmma16(a, bf, acc);
        __syncthreads();
    }

    // epilogue: bias + scatter to head-major f16
    const int gcl = local + wn * 16 + ln;     // col within region
    const int hh  = gcl >> 6;
    const int kko = gcl & 63;
    const float bia = bias[hh * 64 + kko];

    #pragma unroll
    for (int r = 0; r < 8; ++r) {
        int gr = R0 + wm * 16 + r + hi * 8;     // b*2048+s
        int bb = gr >> 11, s = gr & 2047;
        float val = acc[r] + bia;
        if (region == 0) {
            size_t base = ((size_t)(bb * HEADS + hh) * SEQ + s) * DH + kko;
            qh[base] = f2h(val * 0.125f);                 // 1/sqrt(64)
        } else if (region == 1) {
            size_t base = ((size_t)(bb * HEADS + hh) * SEQ + s) * DH + kko;
            kh[base] = f2h(val);
        } else {
            size_t baseT = ((size_t)(bb * HEADS + hh) * DH + kko) * SEQ + s;
            vhT[baseT] = f2h(val);                        // transposed
        }
    }
}

// ---------------------------------------------------------------------------
// Kernel 2: causal flash attention per (b, h, 128 q-rows).
// 8 waves x 16 q rows; s-steps of 32; online softmax; scores stay in VGPRs.
// K and V tiles staged with async global->LDS DMA (one B128 per thread each).
// ---------------------------------------------------------------------------
__global__ __launch_bounds__(256)
void attn_kernel(const unsigned short* __restrict__ qh,
                 const unsigned short* __restrict__ kh,
                 const unsigned short* __restrict__ vhT,
                 unsigned short* __restrict__ zb)
{
    __shared__ __align__(16) unsigned short sK [32 * 64];   // [s][kd]
    __shared__ __align__(16) unsigned short sVT[64 * 32];   // [kk][s]
    __shared__ __align__(16) unsigned short sP [8][16 * 32]; // per-wave P

    const int t = threadIdx.x, lane = t & 31, w = t >> 5;
    const int hi = lane >> 4, ln = lane & 15;

    const int bx   = blockIdx.x;
    const int qblk = bx & 15;
    const int hd   = (bx >> 4) & 15;
    const int bb   = bx >> 8;
    const int Q0   = qblk * 128;
    const int q0   = Q0 + w * 16;

    const size_t headHalves = (size_t)(bb * HEADS + hd) * SEQ * DH;
    const unsigned* gQ  = (const unsigned*)qh + headHalves / 2;
    const char*     gKb = (const char*)kh  + headHalves * 2;
    const char*     gVb = (const char*)vhT + headHalves * 2;

    // per-thread fixed DMA chunk targets/sources
    // K tile: 32 rows x 128 B  -> chunk t: row t>>3, byte (t&7)*16
    // V tile: 64 rows x  64 B  -> chunk t: row t>>2, byte (t&3)*16
    const unsigned ldsK = lds_addr(sK)  + (unsigned)t * 16;
    const unsigned ldsV = lds_addr(sVT) + (unsigned)t * 16;
    const unsigned kRow = t >> 3, kByt = (t & 7) * 16;
    const unsigned vRow = t >> 2, vByt = (t & 3) * 16;

    // Q fragments for both 32-wide d_head chunks; reused for all s-steps.
    FragU aq[2];
    {
        int q = q0 + ln;                         // A layout: lane -> M
        #pragma unroll
        for (int c = 0; c < 2; ++c)
            #pragma unroll
            for (int v = 0; v < 8; ++v) {
                int kd = c * 32 + hi * 8 + ((v < 4) ? 2 * v : 2 * v + 8);
                aq[c].u[v] = gQ[q * 32 + (kd >> 1)];   // row = 32 u32
            }
    }

    float m[8], l[8];
    v8f accz[4];
    v8f vzero = {};
    #pragma unroll
    for (int r = 0; r < 8; ++r) { m[r] = IGN; l[r] = 0.f; }
    #pragma unroll
    for (int tt = 0; tt < 4; ++tt) accz[tt] = vzero;

    const int sEnd = Q0 + 128;                  // causal horizon for this block
    for (int s0 = 0; s0 < sEnd; s0 += 32) {
        // async-DMA K and V tiles into LDS (1x B128 per thread per tile)
        async_ld_b128(ldsK, gKb, (unsigned)(s0 + kRow) * 128u + kByt);
        async_ld_b128(ldsV, gVb, vRow * (SEQ * 2u) + (unsigned)s0 * 2u + vByt);
        wait_async0();
        __syncthreads();

        if (s0 <= q0 + 15) {                    // wave-uniform causal skip
            // ---- scores: two 16-col subtiles, two d_head chunks each ----
            v8f sc[2];
            #pragma unroll
            for (int sub = 0; sub < 2; ++sub) {
                v8f c = {};
                #pragma unroll
                for (int ch = 0; ch < 2; ++ch) {
                    FragU bk;
                    #pragma unroll
                    for (int v = 0; v < 8; ++v) {
                        int kd = ch * 32 + hi * 8 + ((v < 4) ? 2 * v : 2 * v + 8);
                        bk.u[v] = ((const unsigned*)sK)[(sub * 16 + ln) * 32 + (kd >> 1)];
                    }
                    c = wmma16(aq[ch], bk, c);
                }
                sc[sub] = c;
            }
            // ---- mask + online softmax (per C-matrix row) ----
            #pragma unroll
            for (int r = 0; r < 8; ++r) {
                int q  = q0 + r + hi * 8;
                float v0 = ((s0 + ln)      <= q) ? sc[0][r] : IGN;
                float v1 = ((s0 + 16 + ln) <= q) ? sc[1][r] : IGN;
                float mt = rowmax16(fmaxf(v0, v1));
                float mn = fmaxf(m[r], mt);
                float so = __expf(m[r] - mn);
                float p0 = __expf(v0 - mn);
                float p1 = __expf(v1 - mn);
                l[r] = l[r] * so + rowsum16(p0 + p1);
                m[r] = mn;
                sP[w][(r + hi * 8) * 32 + ln]      = f2h(p0);
                sP[w][(r + hi * 8) * 32 + 16 + ln] = f2h(p1);
                #pragma unroll
                for (int tt = 0; tt < 4; ++tt) accz[tt][r] *= so;
            }
            // ---- P re-layout (C -> A fragment) via per-wave LDS ----
            FragU pf;
            #pragma unroll
            for (int v = 0; v < 8; ++v) {
                int scol = hi * 8 + ((v < 4) ? 2 * v : 2 * v + 8);
                pf.u[v] = ((const unsigned*)sP[w])[ln * 16 + (scol >> 1)];
            }
            // ---- P . V into 4 accumulator tiles (16 kk cols each) ----
            #pragma unroll
            for (int tt = 0; tt < 4; ++tt) {
                FragU bv;
                #pragma unroll
                for (int v = 0; v < 8; ++v) {
                    int srow = hi * 8 + ((v < 4) ? 2 * v : 2 * v + 8);
                    bv.u[v] = ((const unsigned*)sVT)[(tt * 16 + ln) * 16 + (srow >> 1)];
                }
                accz[tt] = wmma16(pf, bv, accz[tt]);
            }
        }
        __syncthreads();
    }

    // epilogue: normalize and emit z as f16 [b][s][h*64+kk]
    #pragma unroll
    for (int r = 0; r < 8; ++r) {
        float inv = 1.0f / l[r];
        int q = q0 + r + hi * 8;
        size_t rowb = ((size_t)(bb * SEQ + q)) * DM + hd * DH;
        #pragma unroll
        for (int tt = 0; tt < 4; ++tt)
            zb[rowb + tt * 16 + ln] = f2h(accz[tt][r] * inv);
    }
}

// ---------------------------------------------------------------------------
// Kernel 3: output projection. out[4096,1024] = z * W_O + b_O   (f32 out).
// W_O [h][dh][dm] is contiguous as [1024][1024].
// A tile (z, already f16) staged by async DMA; B tile converted f32->f16.
// ---------------------------------------------------------------------------
__global__ __launch_bounds__(256)
void out_kernel(const unsigned short* __restrict__ zb,
                const float* __restrict__ Wo,
                const float* __restrict__ bo,
                float* __restrict__ out)
{
    __shared__ __align__(16) unsigned short sA[64 * 32];   // z tile [row][k]
    __shared__ __align__(16) unsigned short sB[32 * 32];   // Wo tile [n][k]

    const int t = threadIdx.x, lane = t & 31, w = t >> 5;
    const int wm = w >> 1, wn = w & 1;
    const int hi = lane >> 4, ln = lane & 15;
    const int R0 = blockIdx.y * 64;
    const int C0 = blockIdx.x * 32;

    // A tile: 64 rows x 64 B -> chunk t: row t>>2, byte (t&3)*16
    const unsigned ldsA = lds_addr(sA) + (unsigned)t * 16;
    const unsigned aRow = t >> 2, aByt = (t & 3) * 16;
    const char* gAb = (const char*)zb;

    v8f acc = {};

    for (int kd0 = 0; kd0 < DM; kd0 += 32) {
        // issue async A copy first, overlap with B conversion below
        async_ld_b128(ldsA, gAb,
                      (unsigned)(R0 + aRow) * (DM * 2u) + (unsigned)kd0 * 2u + aByt);
        // stage B: f32 -> f16, transposed [n][k]
        #pragma unroll
        for (int i = 0; i < 8; ++i) {
            int idx = t + 256 * i;
            int dk = idx >> 5, n = idx & 31;
            sB[n * 32 + dk] = f2h(Wo[(size_t)(kd0 + dk) * DM + C0 + n]);
        }
        wait_async0();
        __syncthreads();

        FragU a, bf;
        const unsigned* pA = (const unsigned*)sA;
        const unsigned* pB = (const unsigned*)sB;
        #pragma unroll
        for (int v = 0; v < 8; ++v) {
            int ka = hi * 8 + ((v < 4) ? 2 * v : 2 * v + 8);
            a.u[v]  = pA[(wm * 16 + ln) * 16 + (ka >> 1)];
            int kb = hi * 8 + ((v < 4) ? 2 * v : 2 * v + 8);
            bf.u[v] = pB[(wn * 16 + ln) * 16 + (kb >> 1)];
        }
        acc = wmma16(a, bf, acc);
        __syncthreads();
    }

    const int gc = C0 + wn * 16 + ln;
    const float bia = bo[gc];
    #pragma unroll
    for (int r = 0; r < 8; ++r) {
        int gr = R0 + wm * 16 + r + hi * 8;
        out[(size_t)gr * DM + gc] = acc[r] + bia;
    }
}

// ---------------------------------------------------------------------------
extern "C" void kernel_launch(void* const* d_in, const int* in_sizes, int n_in,
                              void* d_out, int out_size, void* d_ws, size_t ws_size,
                              hipStream_t stream) {
    const float* x  = (const float*)d_in[0];
    const float* Wq = (const float*)d_in[1];
    const float* Wk = (const float*)d_in[2];
    const float* Wv = (const float*)d_in[3];
    const float* Wo = (const float*)d_in[4];
    const float* bq = (const float*)d_in[5];
    const float* bk = (const float*)d_in[6];
    const float* bv = (const float*)d_in[7];
    const float* bo = (const float*)d_in[8];
    float* out = (float*)d_out;

    // workspace carve: q, k, v head-major f16 + z f16  (4 x 8 MB = 32 MB)
    const size_t HBUF = (size_t)2 * HEADS * SEQ * DH;   // 4,194,304 halves
    unsigned short* qh  = (unsigned short*)d_ws;
    unsigned short* kh  = qh + HBUF;
    unsigned short* vhT = kh + HBUF;
    unsigned short* zbf = vhT + HBUF;

    // 1) QKV projection:   grid = (3072/32, 4096/64)
    qkv_kernel<<<dim3(96, 64), 256, 0, stream>>>(x, Wq, Wk, Wv, bq, bk, bv,
                                                 qh, kh, vhT);
    // 2) causal flash attention:  512 blocks = 2 * 16 * (2048/128)
    attn_kernel<<<dim3(512), 256, 0, stream>>>(qh, kh, vhT, zbf);
    // 3) output projection: grid = (1024/32, 4096/64)
    out_kernel<<<dim3(32, 64), 256, 0, stream>>>(zbf, Wo, bo, out);
}
